// Decoder_52424370815716
// MI455X (gfx1250) — matrix operand order repaired and checked by reading, compiled, not run on hardware
//
#include <hip/hip_runtime.h>
#include <hip/hip_bf16.h>

// ---------------- problem constants ----------------
constexpr int B       = 32;
constexpr int T_IN    = 256;
constexpr int T_OUT   = 500;
constexpr int N_MEL   = 80;
constexpr int ENC_DIM = 512;
constexpr int ATT_RNN = 1024;
constexpr int DEC_RNN = 1024;
constexpr int PRENET  = 256;
constexpr int ATT_DIM = 128;
constexpr int LOC_F   = 32;
constexpr int LOC_K   = 31;

constexpr int KA   = PRENET + ENC_DIM + ATT_RNN;   // 1792 : [prenet, ctx, h_att]
constexpr int KD   = ATT_RNN + ENC_DIM + DEC_RNN;  // 2560 : [h_att, ctx, h_dec]
constexpr int KDH  = DEC_RNN + ENC_DIM;            // 1536 : [h_dec, ctx]
constexpr int KP1  = 96;                           // prenet K padded 80 -> 96
constexpr int PROWS = T_OUT * B;                   // 16000 prenet rows

constexpr int DEC_WG = 64;                         // persistent workgroups
constexpr int DEC_TH = 256;

typedef _Float16 half_t;
typedef __attribute__((ext_vector_type(16))) _Float16 v16h;
typedef __attribute__((ext_vector_type(8)))  _Float16 v8h;
typedef __attribute__((ext_vector_type(8)))  float    v8f;

// ---------------- WMMA fragment helpers (16x16x32 f16 -> f32) ----------------
// Per ISA 7.12.2, lane L (n/m = L&15, hi = L>>4) holds:
//   A (16Mx32K row-major):  A[m][hi*8 + 0..7]  and  A[m][16 + hi*8 + 0..7]   (two b128 runs)
//   B (32Kx16N, from W(N,K) row-major): W[n][hi*8 + 0..7]  and  W[n][16 + hi*8 + 0..7]   (two b128 runs)
__device__ __forceinline__ v16h frag_a(const half_t* __restrict__ A, int lda, int lane) {
  int m = lane & 15, hi = lane >> 4;
  const half_t* p = A + m * lda + hi * 8;
  v8h lo = *(const v8h*)(p);
  v8h hh = *(const v8h*)(p + 16);
  return __builtin_shufflevector(lo, hh, 0, 1, 2, 3, 4, 5, 6, 7,
                                         8, 9, 10, 11, 12, 13, 14, 15);
}
__device__ __forceinline__ v16h frag_b(const half_t* __restrict__ W, int ldw, int lane) {
  int n = lane & 15, hi = lane >> 4;
  const half_t* p = W + n * ldw + hi * 8;
  v8h lo = *(const v8h*)(p);
  v8h hh = *(const v8h*)(p + 16);
  return __builtin_shufflevector(lo, hh, 0, 1, 2, 3, 4, 5, 6, 7,
                                         8, 9, 10, 11, 12, 13, 14, 15);
}

__device__ __forceinline__ v8f wmma_tile(const half_t* __restrict__ Ab, int lda,
                                         const half_t* __restrict__ Wb, int ldw,
                                         int K, int lane) {
  v8f acc = {};
  for (int k0 = 0; k0 < K; k0 += 32) {
    v16h af = frag_a(Ab + k0, lda, lane);
    v16h bf = frag_b(Wb + k0, ldw, lane);
    acc = __builtin_amdgcn_wmma_f32_16x16x32_f16(false, af, false, bf,
                                                 (short)0, acc, false, false);
  }
  return acc;
}

// ---------------- prep kernels ----------------
__global__ void k_f32_to_f16(const float* __restrict__ s, half_t* __restrict__ d, int n) {
  for (int i = blockIdx.x * blockDim.x + threadIdx.x; i < n; i += gridDim.x * blockDim.x)
    d[i] = (half_t)s[i];
}
// rows x (ka+kb):  [A(:,0:ka) | B(:,0:kb)]
__global__ void k_concat_rows(const float* __restrict__ a, int ka,
                              const float* __restrict__ b, int kb,
                              half_t* __restrict__ d, int rows) {
  int K = ka + kb;
  int n = rows * K;
  for (int i = blockIdx.x * blockDim.x + threadIdx.x; i < n; i += gridDim.x * blockDim.x) {
    int r = i / K, k = i - r * K;
    float v = (k < ka) ? a[r * ka + k] : b[r * kb + (k - ka)];
    d[i] = (half_t)v;
  }
}
__global__ void k_pad_rows(const float* __restrict__ s, int kin, int kout,
                           half_t* __restrict__ d, int rows) {
  int n = rows * kout;
  for (int i = blockIdx.x * blockDim.x + threadIdx.x; i < n; i += gridDim.x * blockDim.x) {
    int r = i / kout, k = i - r * kout;
    d[i] = (half_t)((k < kin) ? s[r * kin + k] : 0.f);
  }
}
// shifted prenet input: row r = t*B+b, col m<80: t==0 ? 0 : dec_inputs[b,m,t-1]; pad to 96
__global__ void k_prenet_in(const float* __restrict__ dec_inputs, half_t* __restrict__ d) {
  int n = PROWS * KP1;
  for (int i = blockIdx.x * blockDim.x + threadIdx.x; i < n; i += gridDim.x * blockDim.x) {
    int r = i / KP1, m = i - r * KP1;
    int t = r / B, b = r - t * B;
    float v = 0.f;
    if (m < N_MEL && t > 0) v = dec_inputs[(b * N_MEL + m) * T_OUT + (t - 1)];
    d[i] = (half_t)v;
  }
}
__global__ void k_zero_u32(unsigned* p, int n) {
  for (int i = threadIdx.x; i < n; i += blockDim.x) p[i] = 0u;
}

// ---------------- generic WMMA GEMM: C(MxN) = A(MxK) * W(N,K)^T ----------------
__global__ void k_gemm_wmma(const half_t* __restrict__ A, int lda,
                            const half_t* __restrict__ W, int ldw,
                            float* __restrict__ Cf, int ldc,
                            half_t* __restrict__ Ch, int ldch,
                            int M, int N, int K, int relu) {
  int lane = threadIdx.x & 31;
  int wave = (blockIdx.x * blockDim.x + threadIdx.x) >> 5;
  int nwav = (gridDim.x * blockDim.x) >> 5;
  int mt = M >> 4, nt = N >> 4;
  for (int tile = wave; tile < mt * nt; tile += nwav) {
    int tm = (tile / nt) << 4, tn = (tile % nt) << 4;
    v8f acc = wmma_tile(A + tm * lda, lda, W + tn * ldw, ldw, K, lane);
    int hi = lane >> 4, nn = lane & 15;
#pragma unroll
    for (int r = 0; r < 8; ++r) {
      int m = tm + r + hi * 8;
      float v = acc[r];
      if (relu) v = fmaxf(v, 0.f);
      if (Cf) Cf[m * ldc + tn + nn] = v;
      if (Ch) Ch[m * ldch + tn + nn] = (half_t)v;
    }
  }
}

// ---------------- grid barrier (generation counter) ----------------
__device__ __forceinline__ void gbar(unsigned* bar) {
  __syncthreads();
  if (threadIdx.x == 0) {
    __threadfence();
    unsigned g = ((volatile unsigned*)bar)[1];
    unsigned arrived = atomicAdd(&bar[0], 1u);
    if (arrived == (unsigned)(DEC_WG - 1)) {
      bar[0] = 0;
      __threadfence();
      atomicAdd(&bar[1], 1u);
    } else {
      while (((volatile unsigned*)bar)[1] == g) __builtin_amdgcn_s_sleep(1);
    }
    __threadfence();
  }
  __syncthreads();
}

__device__ __forceinline__ float sigf(float x) { return 1.f / (1.f + __expf(-x)); }

// ---------------- persistent decoder ----------------
__global__ void __launch_bounds__(DEC_TH)
k_decoder(const half_t* __restrict__ pre_out,   // (T_OUT*B, 256) f16
          const half_t* __restrict__ wcat_att,  // (4096, 1792)
          const half_t* __restrict__ wcat_dec,  // (4096, 2560)
          const half_t* __restrict__ w_query,   // (128, 1024)
          const half_t* __restrict__ w_proj,    // (80, 1536)
          const float*  __restrict__ att_b_ih, const float* __restrict__ att_b_hh,
          const float*  __restrict__ dec_b_ih, const float* __restrict__ dec_b_hh,
          const float*  __restrict__ proj_b,
          const float*  __restrict__ gate_w,   const float* __restrict__ gate_b,
          const float*  __restrict__ v_w,
          const float*  __restrict__ loc_conv_w,  // (32,2,31)
          const float*  __restrict__ loc_dense_w, // (128,32)
          const float*  __restrict__ memory,      // (B,T_IN,512) f32
          const int*    __restrict__ mem_len,
          const float*  __restrict__ pm,          // (B*T_IN,128)
          float* gates, float* h_att, float* c_att, float* h_dec, float* c_dec,
          float* ctx, float* aw, float* aw_cum, float* pq, float* convbuf, float* energies,
          half_t* xa, half_t* xq, half_t* xd, half_t* dhac,
          unsigned* bar,
          float* out_mel, float* out_gate, float* out_align) {
  const int tid  = blockIdx.x * blockDim.x + threadIdx.x;
  const int nthr = gridDim.x * blockDim.x;
  const int lane = threadIdx.x & 31;
  const int wave = tid >> 5;
  const int nwav = nthr >> 5;
  __shared__ float red[T_IN];
  __shared__ float wsh[T_IN];

  // -------- zero-init recurrent state (fresh every call) --------
  for (int i = tid; i < B * ATT_RNN; i += nthr) { h_att[i] = 0.f; c_att[i] = 0.f; h_dec[i] = 0.f; c_dec[i] = 0.f; xq[i] = (half_t)0.f; }
  for (int i = tid; i < B * ENC_DIM; i += nthr) ctx[i] = 0.f;
  for (int i = tid; i < B * T_IN; i += nthr) { aw[i] = 0.f; aw_cum[i] = 0.f; }
  gbar(bar);

  for (int t = 0; t < T_OUT; ++t) {
    // ---- P1: build A for attention LSTM: [prenet_t | ctx | h_att] (f16) ----
    for (int i = tid; i < B * KA; i += nthr) {
      int b = i / KA, k = i - b * KA;
      half_t v;
      if (k < PRENET)              v = pre_out[(t * B + b) * PRENET + k];
      else if (k < PRENET+ENC_DIM) v = (half_t)ctx[b * ENC_DIM + (k - PRENET)];
      else                         v = (half_t)h_att[b * ATT_RNN + (k - PRENET - ENC_DIM)];
      xa[i] = v;
    }
    gbar(bar);
    // ---- P2: attention LSTM gates GEMM  (32 x 4096, K=1792) ----
    for (int tile = wave; tile < 2 * (4 * ATT_RNN / 16); tile += nwav) {
      int tm = (tile >> 8) << 4, tn = (tile & 255) << 4;
      v8f acc = wmma_tile(xa + tm * KA, KA, wcat_att + tn * KA, KA, KA, lane);
      int hi = lane >> 4, nn = lane & 15;
#pragma unroll
      for (int r = 0; r < 8; ++r) {
        int m = tm + r + hi * 8, n = tn + nn;
        gates[m * 4096 + n] = acc[r] + att_b_ih[n] + att_b_hh[n];
      }
    }
    gbar(bar);
    // ---- P3: attention LSTM pointwise ----
    for (int i = tid; i < B * ATT_RNN; i += nthr) {
      int b = i >> 10, u = i & 1023;
      const float* g = gates + b * 4096;
      float c = sigf(g[1024 + u]) * c_att[i] + sigf(g[u]) * tanhf(g[2048 + u]);
      float h = sigf(g[3072 + u]) * tanhf(c);
      c_att[i] = c; h_att[i] = h; xq[i] = (half_t)h;
    }
    gbar(bar);
    // ---- P4: query projection GEMM (32x128, K=1024) + location conv ----
    for (int tile = wave; tile < 2 * (ATT_DIM / 16); tile += nwav) {
      int tm = (tile >> 3) << 4, tn = (tile & 7) << 4;
      v8f acc = wmma_tile(xq + tm * ATT_RNN, ATT_RNN, w_query + tn * ATT_RNN, ATT_RNN, ATT_RNN, lane);
      int hi = lane >> 4, nn = lane & 15;
#pragma unroll
      for (int r = 0; r < 8; ++r) pq[(tm + r + hi * 8) * ATT_DIM + tn + nn] = acc[r];
    }
    for (int i = tid; i < B * LOC_F * T_IN; i += nthr) {
      int b = i / (LOC_F * T_IN), rem = i - b * (LOC_F * T_IN);
      int f = rem / T_IN, tp = rem - f * T_IN;
      float s = 0.f;
#pragma unroll 1
      for (int kk = 0; kk < LOC_K; ++kk) {
        int src = tp + kk - (LOC_K - 1) / 2;
        if (src >= 0 && src < T_IN) {
          s += loc_conv_w[(f * 2 + 0) * LOC_K + kk] * aw[b * T_IN + src];
          s += loc_conv_w[(f * 2 + 1) * LOC_K + kk] * aw_cum[b * T_IN + src];
        }
      }
      convbuf[i] = s;
    }
    gbar(bar);
    // ---- P5: energies + length mask ----
    for (int i = tid; i < B * T_IN; i += nthr) {
      int b = i >> 8, tp = i & 255;
      const float* pqb = pq + b * ATT_DIM;
      const float* pmb = pm + (b * T_IN + tp) * ATT_DIM;
      const float* cb  = convbuf + b * (LOC_F * T_IN) + tp;
      float e = 0.f;
      for (int a = 0; a < ATT_DIM; ++a) {
        float ld = 0.f;
#pragma unroll
        for (int f = 0; f < LOC_F; ++f) ld += cb[f * T_IN] * loc_dense_w[a * LOC_F + f];
        e += v_w[a] * tanhf(pqb[a] + ld + pmb[a]);
      }
      if (tp >= mem_len[b]) e = -1e30f;
      energies[i] = e;
    }
    gbar(bar);
    // ---- P6: per-batch softmax + context (block b handles batch b) ----
    if (blockIdx.x < B) {
      int b = blockIdx.x, tp = threadIdx.x;
      float e = energies[b * T_IN + tp];
      red[tp] = e; __syncthreads();
      for (int s = T_IN / 2; s > 0; s >>= 1) { if (tp < s) red[tp] = fmaxf(red[tp], red[tp + s]); __syncthreads(); }
      float mx = red[0]; __syncthreads();
      float ex = __expf(e - mx);
      red[tp] = ex; __syncthreads();
      for (int s = T_IN / 2; s > 0; s >>= 1) { if (tp < s) red[tp] += red[tp + s]; __syncthreads(); }
      float w = ex / red[0];
      aw[b * T_IN + tp] = w;
      aw_cum[b * T_IN + tp] += w;
      wsh[tp] = w; __syncthreads();
      for (int d = tp; d < ENC_DIM; d += T_IN) {
        float s = 0.f;
        for (int t2 = 0; t2 < T_IN; ++t2) s += wsh[t2] * memory[(b * T_IN + t2) * ENC_DIM + d];
        ctx[b * ENC_DIM + d] = s;
      }
    }
    gbar(bar);
    // ---- P7: build A for decoder LSTM [h_att | ctx | h_dec], dump alignments ----
    for (int i = tid; i < B * KD; i += nthr) {
      int b = i / KD, k = i - b * KD;
      half_t v;
      if (k < ATT_RNN)              v = xq[b * ATT_RNN + k];
      else if (k < ATT_RNN+ENC_DIM) v = (half_t)ctx[b * ENC_DIM + (k - ATT_RNN)];
      else                          v = (half_t)h_dec[b * DEC_RNN + (k - ATT_RNN - ENC_DIM)];
      xd[i] = v;
    }
    for (int i = tid; i < B * T_IN; i += nthr) {
      int b = i >> 8, tp = i & 255;
      out_align[(b * T_OUT + t) * T_IN + tp] = aw[i];
    }
    gbar(bar);
    // ---- P8: decoder LSTM gates GEMM (32 x 4096, K=2560) ----
    for (int tile = wave; tile < 2 * (4 * DEC_RNN / 16); tile += nwav) {
      int tm = (tile >> 8) << 4, tn = (tile & 255) << 4;
      v8f acc = wmma_tile(xd + tm * KD, KD, wcat_dec + tn * KD, KD, KD, lane);
      int hi = lane >> 4, nn = lane & 15;
#pragma unroll
      for (int r = 0; r < 8; ++r) {
        int m = tm + r + hi * 8, n = tn + nn;
        gates[m * 4096 + n] = acc[r] + dec_b_ih[n] + dec_b_hh[n];
      }
    }
    gbar(bar);
    // ---- P9: decoder LSTM pointwise + build dhac = [h_dec | ctx] ----
    for (int i = tid; i < B * DEC_RNN; i += nthr) {
      int b = i >> 10, u = i & 1023;
      const float* g = gates + b * 4096;
      float c = sigf(g[1024 + u]) * c_dec[i] + sigf(g[u]) * tanhf(g[2048 + u]);
      float h = sigf(g[3072 + u]) * tanhf(c);
      c_dec[i] = c; h_dec[i] = h;
      dhac[b * KDH + u] = (half_t)h;
    }
    for (int i = tid; i < B * ENC_DIM; i += nthr) {
      int b = i >> 9, d = i & 511;
      dhac[b * KDH + DEC_RNN + d] = (half_t)ctx[i];
    }
    gbar(bar);
    // ---- P10: mel projection GEMM (32 x 80, K=1536) + gate dot ----
    for (int tile = wave; tile < 2 * (N_MEL / 16); tile += nwav) {
      int tm = (tile / 5) << 4, tn = (tile % 5) << 4;
      v8f acc = wmma_tile(dhac + tm * KDH, KDH, w_proj + tn * KDH, KDH, KDH, lane);
      int hi = lane >> 4, nn = lane & 15;
#pragma unroll
      for (int r = 0; r < 8; ++r) {
        int b = tm + r + hi * 8, n = tn + nn;
        out_mel[(b * N_MEL + n) * T_OUT + t] = acc[r] + proj_b[n];
      }
    }
    for (int b = wave; b < B; b += nwav) {
      float s = 0.f;
      for (int k = lane; k < KDH; k += 32) s += (float)dhac[b * KDH + k] * gate_w[k];
      for (int off = 16; off > 0; off >>= 1) s += __shfl_down(s, off, 32);
      if (lane == 0) out_gate[b * T_OUT + t] = s + gate_b[0];
    }
    gbar(bar);
  }
}

// ---------------- host side ----------------
static inline size_t al256(size_t x) { return (x + 255) & ~(size_t)255; }

extern "C" void kernel_launch(void* const* d_in, const int* in_sizes, int n_in,
                              void* d_out, int out_size, void* d_ws, size_t ws_size,
                              hipStream_t stream) {
  const float* memory      = (const float*)d_in[0];
  const float* dec_inputs  = (const float*)d_in[1];
  const int*   mem_len     = (const int*)  d_in[2];
  const float* prenet_w1   = (const float*)d_in[3];
  const float* prenet_w2   = (const float*)d_in[4];
  const float* att_w_ih    = (const float*)d_in[5];
  const float* att_w_hh    = (const float*)d_in[6];
  const float* att_b_ih    = (const float*)d_in[7];
  const float* att_b_hh    = (const float*)d_in[8];
  const float* query_w     = (const float*)d_in[9];
  const float* memory_w    = (const float*)d_in[10];
  const float* v_w         = (const float*)d_in[11];
  const float* loc_conv_w  = (const float*)d_in[12];
  const float* loc_dense_w = (const float*)d_in[13];
  const float* dec_w_ih    = (const float*)d_in[14];
  const float* dec_w_hh    = (const float*)d_in[15];
  const float* dec_b_ih    = (const float*)d_in[16];
  const float* dec_b_hh    = (const float*)d_in[17];
  const float* proj_w      = (const float*)d_in[18];
  const float* proj_b      = (const float*)d_in[19];
  const float* gate_w      = (const float*)d_in[20];
  const float* gate_b      = (const float*)d_in[21];

  char* ws = (char*)d_ws;
  size_t cur = 0;
  auto takeH = [&](size_t n) { half_t* p = (half_t*)(ws + cur); cur = al256(cur + n * sizeof(half_t)); return p; };
  auto takeF = [&](size_t n) { float*  p = (float*) (ws + cur); cur = al256(cur + n * sizeof(float));  return p; };

  half_t* wcat_att = takeH((size_t)4096 * KA);
  half_t* wcat_dec = takeH((size_t)4096 * KD);
  half_t* w_query  = takeH((size_t)ATT_DIM * ATT_RNN);
  half_t* w_mem    = takeH((size_t)ATT_DIM * ENC_DIM);
  half_t* w_pre1   = takeH((size_t)PRENET * KP1);
  half_t* w_pre2   = takeH((size_t)PRENET * PRENET);
  half_t* w_proj   = takeH((size_t)N_MEL * KDH);
  half_t* mem_h    = takeH((size_t)B * T_IN * ENC_DIM);
  half_t* pre_in   = takeH((size_t)PROWS * KP1);
  half_t* pre_mid  = takeH((size_t)PROWS * PRENET);
  half_t* pre_out  = takeH((size_t)PROWS * PRENET);
  half_t* xa       = takeH((size_t)B * KA);
  half_t* xq       = takeH((size_t)B * ATT_RNN);
  half_t* xd       = takeH((size_t)B * KD);
  half_t* dhac     = takeH((size_t)B * KDH);
  float* pm        = takeF((size_t)B * T_IN * ATT_DIM);
  float* gates     = takeF((size_t)B * 4096);
  float* h_att     = takeF((size_t)B * ATT_RNN);
  float* c_att     = takeF((size_t)B * ATT_RNN);
  float* h_dec     = takeF((size_t)B * DEC_RNN);
  float* c_dec     = takeF((size_t)B * DEC_RNN);
  float* ctx       = takeF((size_t)B * ENC_DIM);
  float* aw        = takeF((size_t)B * T_IN);
  float* aw_cum    = takeF((size_t)B * T_IN);
  float* pq        = takeF((size_t)B * ATT_DIM);
  float* convbuf   = takeF((size_t)B * LOC_F * T_IN);
  float* energies  = takeF((size_t)B * T_IN);
  unsigned* bar    = (unsigned*)(ws + cur); cur = al256(cur + 2 * sizeof(unsigned));
  (void)ws_size; (void)n_in; (void)in_sizes; (void)out_size;

  float* out_mel   = (float*)d_out;
  float* out_gate  = out_mel + (size_t)B * N_MEL * T_OUT;
  float* out_align = out_gate + (size_t)B * T_OUT;

  dim3 pb(256);
  dim3 pg(256);
  // weight conversion / concatenation
  hipLaunchKernelGGL(k_concat_rows, pg, pb, 0, stream, att_w_ih, PRENET + ENC_DIM, att_w_hh, ATT_RNN, wcat_att, 4 * ATT_RNN);
  hipLaunchKernelGGL(k_concat_rows, pg, pb, 0, stream, dec_w_ih, ATT_RNN + ENC_DIM, dec_w_hh, DEC_RNN, wcat_dec, 4 * DEC_RNN);
  hipLaunchKernelGGL(k_f32_to_f16, pg, pb, 0, stream, query_w, w_query, ATT_DIM * ATT_RNN);
  hipLaunchKernelGGL(k_f32_to_f16, pg, pb, 0, stream, memory_w, w_mem, ATT_DIM * ENC_DIM);
  hipLaunchKernelGGL(k_f32_to_f16, pg, pb, 0, stream, prenet_w2, w_pre2, PRENET * PRENET);
  hipLaunchKernelGGL(k_f32_to_f16, pg, pb, 0, stream, proj_w, w_proj, N_MEL * KDH);
  hipLaunchKernelGGL(k_f32_to_f16, pg, pb, 0, stream, memory, mem_h, B * T_IN * ENC_DIM);
  hipLaunchKernelGGL(k_pad_rows, pg, pb, 0, stream, prenet_w1, N_MEL, KP1, w_pre1, PRENET);
  hipLaunchKernelGGL(k_prenet_in, pg, pb, 0, stream, dec_inputs, pre_in);

  // timestep-parallel GEMMs (WMMA)
  dim3 gg(256), gb(256);
  hipLaunchKernelGGL(k_gemm_wmma, gg, gb, 0, stream, pre_in, KP1, w_pre1, KP1,
                     (float*)nullptr, 0, pre_mid, PRENET, PROWS, PRENET, KP1, 1);
  hipLaunchKernelGGL(k_gemm_wmma, gg, gb, 0, stream, pre_mid, PRENET, w_pre2, PRENET,
                     (float*)nullptr, 0, pre_out, PRENET, PROWS, PRENET, PRENET, 1);
  hipLaunchKernelGGL(k_gemm_wmma, gg, gb, 0, stream, mem_h, ENC_DIM, w_mem, ENC_DIM,
                     pm, ATT_DIM, (half_t*)nullptr, 0, B * T_IN, ATT_DIM, ENC_DIM, 0);

  hipLaunchKernelGGL(k_zero_u32, dim3(1), dim3(64), 0, stream, bar, 2);

  // persistent sequential decoder
  hipLaunchKernelGGL(k_decoder, dim3(DEC_WG), dim3(DEC_TH), 0, stream,
                     pre_out, wcat_att, wcat_dec, w_query, w_proj,
                     att_b_ih, att_b_hh, dec_b_ih, dec_b_hh, proj_b,
                     gate_w, gate_b, v_w, loc_conv_w, loc_dense_w,
                     memory, mem_len, pm,
                     gates, h_att, c_att, h_dec, c_dec,
                     ctx, aw, aw_cum, pq, convbuf, energies,
                     xa, xq, xd, dhac, bar,
                     out_mel, out_gate, out_align);
}
